// LinearLayer_15874199126079
// MI455X (gfx1250) — matrix operand order repaired and checked by reading, not compiled, not run on hardware
//
#include <hip/hip_runtime.h>

// ---------------------------------------------------------------------------
// y = x @ W^T + b  on gfx1250 via bf16x3 split-precision WMMA.
// Phase 1: split f32 -> (bf16 hi, bf16 residual) arrays in workspace.
// Phase 2: GEMM, double-buffered async global->LDS staging (ASYNCcnt DMA)
//          overlapped with v_wmma_f32_16x16x32_bf16 compute.
// Fallback (workspace too small): fused convert-in-loop kernel.
// ---------------------------------------------------------------------------

typedef __attribute__((ext_vector_type(16))) __bf16 v16bf;
typedef __attribute__((ext_vector_type(8)))  __bf16 v8bf;
typedef __attribute__((ext_vector_type(8)))  float  v8f;

#define M_TOTAL 8192
#define N_TOTAL 4096
#define K_TOTAL 4096

// main GEMM tiling: 128x256 block tile, 8 waves, 64x64 wave tile, K-step 32
#define BM 128
#define BN 256
#define BK 32
#define LDT 40   // padded LDS row stride (80B) -> conflict-free b128 access

// ---------------- bf16 helpers ----------------
__device__ __forceinline__ unsigned short f32_bf16_rne(float f) {
    unsigned u = __builtin_bit_cast(unsigned, f);
    unsigned r = u + 0x7FFFu + ((u >> 16) & 1u);
    return (unsigned short)(r >> 16);
}
__device__ __forceinline__ float bf16_f32(unsigned short s) {
    return __builtin_bit_cast(float, ((unsigned)s) << 16);
}
__device__ __forceinline__ v8f wmma_bf16(v16bf a, v16bf b, v8f c) {
    return __builtin_amdgcn_wmma_f32_16x16x32_bf16(false, a, false, b,
                                                   (short)0, c, false, false);
}

// A-fragment (16x32): lane holds row M=lane%16; kh=lane/16 selects K chunks
// {kh*8..+8} (VGPR0-3) and {16+kh*8..+8} (VGPR4-7).
__device__ __forceinline__ v16bf frag_a(const unsigned short* s, int row, int kh) {
    const v8bf c0 = *reinterpret_cast<const v8bf*>(s + row * LDT + kh * 8);
    const v8bf c1 = *reinterpret_cast<const v8bf*>(s + row * LDT + kh * 8 + 16);
    return __builtin_shufflevector(c0, c1, 0,1,2,3,4,5,6,7,8,9,10,11,12,13,14,15);
}
// B-fragment (32x16): lane holds column N=lane%16, K chunks {kh*8..+8} and {16+kh*8..+8}.
__device__ __forceinline__ v16bf frag_b(const unsigned short* s, int col, int kh) {
    const v8bf c0 = *reinterpret_cast<const v8bf*>(s + col * LDT + kh * 8);
    const v8bf c1 = *reinterpret_cast<const v8bf*>(s + col * LDT + kh * 8 + 16);
    return __builtin_shufflevector(c0, c1, 0,1,2,3,4,5,6,7,8,9,10,11,12,13,14,15);
}

// ---------------- phase 1: f32 -> bf16 hi/lo split ----------------
__global__ __launch_bounds__(256)
void split_bf16_kernel(const float* __restrict__ src,
                       unsigned short* __restrict__ hi,
                       unsigned short* __restrict__ lo, int n4) {
    int i = blockIdx.x * blockDim.x + threadIdx.x;
    if (i >= n4) return;
    float f[4];
    *reinterpret_cast<float4*>(f) = reinterpret_cast<const float4*>(src)[i];
    unsigned hp[2], lp[2];
#pragma unroll
    for (int j = 0; j < 2; ++j) {
        unsigned short h0 = f32_bf16_rne(f[2 * j]);
        unsigned short h1 = f32_bf16_rne(f[2 * j + 1]);
        unsigned short l0 = f32_bf16_rne(f[2 * j] - bf16_f32(h0));
        unsigned short l1 = f32_bf16_rne(f[2 * j + 1] - bf16_f32(h1));
        hp[j] = (unsigned)h0 | ((unsigned)h1 << 16);
        lp[j] = (unsigned)l0 | ((unsigned)l1 << 16);
    }
    reinterpret_cast<uint2*>(hi)[i] = *reinterpret_cast<const uint2*>(hp);
    reinterpret_cast<uint2*>(lo)[i] = *reinterpret_cast<const uint2*>(lp);
}

// ---------------- async global -> LDS copy (16B per lane) ----------------
__device__ __forceinline__ void async_copy_b128(const unsigned short* g,
                                                unsigned short* l) {
    unsigned lds = (unsigned)(unsigned long long)(const void*)l;  // LDS offset
    asm volatile("global_load_async_to_lds_b128 %0, %1, off"
                 :: "v"(lds), "v"(g) : "memory");
}
// wait until at most the newest 12 async ops (next tile) remain outstanding
__device__ __forceinline__ void wait_async_le12() {
    asm volatile("s_wait_asynccnt 0xc" ::: "memory");
}
__device__ __forceinline__ void wait_async0() {
    asm volatile("s_wait_asynccnt 0x0" ::: "memory");
}

// issue one tile's 12 async b128 copies (per thread)
__device__ __forceinline__ void issue_tile(const unsigned short* gAhi,
                                           const unsigned short* gAlo,
                                           const unsigned short* gBhi,
                                           const unsigned short* gBlo,
                                           unsigned short* lAhi, unsigned short* lAlo,
                                           unsigned short* lBhi, unsigned short* lBlo) {
    async_copy_b128(gAhi,     lAhi);
    async_copy_b128(gAhi + 8, lAhi + 8);
    async_copy_b128(gAlo,     lAlo);
    async_copy_b128(gAlo + 8, lAlo + 8);
#pragma unroll
    for (int c = 0; c < 4; ++c) {
        async_copy_b128(gBhi + 8 * c, lBhi + 8 * c);
        async_copy_b128(gBlo + 8 * c, lBlo + 8 * c);
    }
}

// ---------------- phase 2: main GEMM (double-buffered pipeline) ----------
__global__ __launch_bounds__(256, 1)
void gemm_bf16x3_wmma_async(const unsigned short* __restrict__ Xhi,
                            const unsigned short* __restrict__ Xlo,
                            const unsigned short* __restrict__ Whi,
                            const unsigned short* __restrict__ Wlo,
                            const float* __restrict__ bias,
                            float* __restrict__ out) {
    __shared__ __align__(16) unsigned short sAhi[2][BM * LDT];
    __shared__ __align__(16) unsigned short sAlo[2][BM * LDT];
    __shared__ __align__(16) unsigned short sBhi[2][BN * LDT];
    __shared__ __align__(16) unsigned short sBlo[2][BN * LDT];

    const int tid   = threadIdx.x;
    const int lane  = tid & 31;
    const int wave  = tid >> 5;      // 8 waves
    const int wm    = wave & 1;      // 2 x 64 rows
    const int wn    = wave >> 1;     // 4 x 64 cols
    const int lrow  = lane & 15;
    const int khalf = lane >> 4;

    // staging assignment
    const int arow = tid >> 1;           // A: 128 rows, 16 elems/thread
    const int acol = (tid & 1) * 16;
    const int brow = tid;                // B: 256 rows, 32 elems/thread
    const int aoff = arow * LDT + acol;
    const int boff = brow * LDT;

    const unsigned short* gAhi = Xhi + (size_t)(blockIdx.y * BM + arow) * K_TOTAL + acol;
    const unsigned short* gAlo = Xlo + (size_t)(blockIdx.y * BM + arow) * K_TOTAL + acol;
    const unsigned short* gBhi = Whi + (size_t)(blockIdx.x * BN + brow) * K_TOTAL;
    const unsigned short* gBlo = Wlo + (size_t)(blockIdx.x * BN + brow) * K_TOTAL;

    v8f acc[4][4] = {};

    const int NIT = K_TOTAL / BK;    // 128

    // prologue: start DMA for tile 0 into buffer 0
    issue_tile(gAhi, gAlo, gBhi, gBlo,
               &sAhi[0][aoff], &sAlo[0][aoff], &sBhi[0][boff], &sBlo[0][boff]);
    gAhi += BK; gAlo += BK; gBhi += BK; gBlo += BK;

    for (int i = 0; i < NIT; ++i) {
        const int cur = i & 1;
        const int nxt = cur ^ 1;

        if (i + 1 < NIT) {
            // start DMA for tile i+1 while tile i finishes / is computed
            issue_tile(gAhi, gAlo, gBhi, gBlo,
                       &sAhi[nxt][aoff], &sAlo[nxt][aoff],
                       &sBhi[nxt][boff], &sBlo[nxt][boff]);
            gAhi += BK; gAlo += BK; gBhi += BK; gBlo += BK;
            wait_async_le12();   // in-order: tile i's 12 copies are complete
        } else {
            wait_async0();
        }
        __syncthreads();         // tile i visible to all waves

        const unsigned short* cAhi = sAhi[cur];
        const unsigned short* cAlo = sAlo[cur];
        const unsigned short* cBhi = sBhi[cur];
        const unsigned short* cBlo = sBlo[cur];

        v16bf ah[4], al[4];
#pragma unroll
        for (int mi = 0; mi < 4; ++mi) {
            const int r = wm * 64 + mi * 16 + lrow;
            ah[mi] = frag_a(cAhi, r, khalf);
            al[mi] = frag_a(cAlo, r, khalf);
        }
#pragma unroll
        for (int ni = 0; ni < 4; ++ni) {
            const int o = wn * 64 + ni * 16 + lrow;
            const v16bf bh = frag_b(cBhi, o, khalf);
            const v16bf bl = frag_b(cBlo, o, khalf);
#pragma unroll
            for (int mi = 0; mi < 4; ++mi) {
                acc[mi][ni] = wmma_bf16(ah[mi], bh, acc[mi][ni]);  // hi*hi
                acc[mi][ni] = wmma_bf16(al[mi], bh, acc[mi][ni]);  // lo*hi
                acc[mi][ni] = wmma_bf16(ah[mi], bl, acc[mi][ni]);  // hi*lo
            }
        }
        __syncthreads();         // all waves done with buffer `cur` before it
                                 // is overwritten by iteration i+1's DMA
    }

    // epilogue: C/D layout  M = r + 8*(lane>>4), N = lane&15
#pragma unroll
    for (int mi = 0; mi < 4; ++mi) {
        const int row0 = blockIdx.y * BM + wm * 64 + mi * 16 + khalf * 8;
#pragma unroll
        for (int ni = 0; ni < 4; ++ni) {
            const int col = blockIdx.x * BN + wn * 64 + ni * 16 + lrow;
            const float bv = bias[col];
            const v8f c = acc[mi][ni];
#pragma unroll
            for (int r = 0; r < 8; ++r)
                out[(size_t)(row0 + r) * N_TOTAL + col] = c[r] + bv;
        }
    }
}

// ---------------- fallback: fused convert-in-loop kernel (128x128) --------
__device__ __forceinline__ void cvt_split_store(const float4 r[4],
                                                unsigned short* __restrict__ shi,
                                                unsigned short* __restrict__ slo,
                                                int row, int col0) {
    float f[16];
#pragma unroll
    for (int j = 0; j < 4; ++j) *reinterpret_cast<float4*>(f + 4 * j) = r[j];
#pragma unroll
    for (int c = 0; c < 2; ++c) {
        unsigned hp[4], lp[4];
#pragma unroll
        for (int i = 0; i < 4; ++i) {
            float f0 = f[c * 8 + 2 * i], f1 = f[c * 8 + 2 * i + 1];
            unsigned short h0 = f32_bf16_rne(f0);
            unsigned short h1 = f32_bf16_rne(f1);
            unsigned short l0 = f32_bf16_rne(f0 - bf16_f32(h0));
            unsigned short l1 = f32_bf16_rne(f1 - bf16_f32(h1));
            hp[i] = (unsigned)h0 | ((unsigned)h1 << 16);
            lp[i] = (unsigned)l0 | ((unsigned)l1 << 16);
        }
        *reinterpret_cast<uint4*>(shi + row * LDT + col0 + c * 8) =
            *reinterpret_cast<const uint4*>(hp);
        *reinterpret_cast<uint4*>(slo + row * LDT + col0 + c * 8) =
            *reinterpret_cast<const uint4*>(lp);
    }
}

__global__ __launch_bounds__(256)
void linear_bf16x3_wmma_fused(const float* __restrict__ A,
                              const float* __restrict__ W,
                              const float* __restrict__ bias,
                              float* __restrict__ out) {
    __shared__ __align__(16) unsigned short sAhi[128 * LDT];
    __shared__ __align__(16) unsigned short sAlo[128 * LDT];
    __shared__ __align__(16) unsigned short sBhi[128 * LDT];
    __shared__ __align__(16) unsigned short sBlo[128 * LDT];

    const int tid = threadIdx.x, lane = tid & 31, wave = tid >> 5;
    const int wm = wave & 3, wn = wave >> 2;
    const int lrow = lane & 15, khalf = lane >> 4;
    const int srow = tid >> 1, scol = (tid & 1) * 16;

    const float* gA = A + (size_t)(blockIdx.y * 128 + srow) * K_TOTAL + scol;
    const float* gW = W + (size_t)(blockIdx.x * 128 + srow) * K_TOTAL + scol;

    v8f acc[2][4] = {};
    for (int k0 = 0; k0 < K_TOTAL; k0 += BK) {
        float4 areg[4], breg[4];
#pragma unroll
        for (int j = 0; j < 4; ++j) {
            areg[j] = *reinterpret_cast<const float4*>(gA + 4 * j);
            breg[j] = *reinterpret_cast<const float4*>(gW + 4 * j);
        }
        __builtin_prefetch(gA + BK, 0, 0);
        __builtin_prefetch(gW + BK, 0, 0);
        gA += BK; gW += BK;
        __syncthreads();
        cvt_split_store(areg, sAhi, sAlo, srow, scol);
        cvt_split_store(breg, sBhi, sBlo, srow, scol);
        __syncthreads();

        v16bf ah[2], al[2];
#pragma unroll
        for (int mi = 0; mi < 2; ++mi) {
            const int r = wm * 32 + mi * 16 + lrow;
            ah[mi] = frag_a(sAhi, r, khalf);
            al[mi] = frag_a(sAlo, r, khalf);
        }
#pragma unroll
        for (int ni = 0; ni < 4; ++ni) {
            const int o = wn * 64 + ni * 16 + lrow;
            const v16bf bh = frag_b(sBhi, o, khalf);
            const v16bf bl = frag_b(sBlo, o, khalf);
#pragma unroll
            for (int mi = 0; mi < 2; ++mi) {
                acc[mi][ni] = wmma_bf16(ah[mi], bh, acc[mi][ni]);
                acc[mi][ni] = wmma_bf16(al[mi], bh, acc[mi][ni]);
                acc[mi][ni] = wmma_bf16(ah[mi], bl, acc[mi][ni]);
            }
        }
    }
#pragma unroll
    for (int mi = 0; mi < 2; ++mi) {
        const int row0 = blockIdx.y * 128 + wm * 32 + mi * 16 + khalf * 8;
#pragma unroll
        for (int ni = 0; ni < 4; ++ni) {
            const int col = blockIdx.x * 128 + wn * 64 + ni * 16 + lrow;
            const float bv = bias[col];
            const v8f c = acc[mi][ni];
#pragma unroll
            for (int r = 0; r < 8; ++r)
                out[(size_t)(row0 + r) * N_TOTAL + col] = c[r] + bv;
        }
    }
}

// ---------------- launch ----------------
extern "C" void kernel_launch(void* const* d_in, const int* in_sizes, int n_in,
                              void* d_out, int out_size, void* d_ws, size_t ws_size,
                              hipStream_t stream) {
    (void)in_sizes; (void)n_in; (void)out_size;
    const float* inputs  = (const float*)d_in[0];  // [8192, 4096]
    const float* weights = (const float*)d_in[1];  // [4096, 4096]
    const float* bias    = (const float*)d_in[2];  // [4096]
    float* out = (float*)d_out;

    const size_t xe = (size_t)M_TOTAL * K_TOTAL;   // X elements
    const size_t we = (size_t)N_TOTAL * K_TOTAL;   // W elements
    const size_t need = (2 * xe + 2 * we) * sizeof(unsigned short);  // 192 MB

    if (ws_size >= need) {
        unsigned short* Xhi = (unsigned short*)d_ws;
        unsigned short* Xlo = Xhi + xe;
        unsigned short* Whi = Xlo + xe;
        unsigned short* Wlo = Whi + we;

        const int n4x = (int)(xe / 4), n4w = (int)(we / 4);
        split_bf16_kernel<<<(n4x + 255) / 256, 256, 0, stream>>>(inputs, Xhi, Xlo, n4x);
        split_bf16_kernel<<<(n4w + 255) / 256, 256, 0, stream>>>(weights, Whi, Wlo, n4w);

        dim3 grid(N_TOTAL / BN, M_TOTAL / BM);     // (16, 64)
        gemm_bf16x3_wmma_async<<<grid, 256, 0, stream>>>(Xhi, Xlo, Whi, Wlo, bias, out);
    } else {
        dim3 grid(N_TOTAL / 128, M_TOTAL / 128);   // (32, 64)
        linear_bf16x3_wmma_fused<<<grid, 256, 0, stream>>>(inputs, weights, bias, out);
    }
}
